// MultiHeadSelfAttention_15994458211157
// MI455X (gfx1250) — matrix operand order repaired and checked by reading, not compiled, not run on hardware
//
#include <hip/hip_runtime.h>
#include <hip/hip_bf16.h>
#include <math.h>

// ---------------------------------------------------------------------------
// MultiHeadSelfAttention forward on gfx1250 (MI455X).
// All matmuls via V_WMMA_F32_16X16X32_F16 (f16 in, f32 accumulate).
// Flash-style attention: the (B,H,N,N) score matrix never touches HBM.
// Tile pipeline: double-buffered LDS, non-transposed tiles filled with
// GLOBAL_LOAD_ASYNC_TO_LDS_B128 (ASYNCcnt) issued one iteration ahead;
// transposed tiles fetched to VGPRs early and stored to LDS after compute.
// ---------------------------------------------------------------------------

typedef __attribute__((ext_vector_type(16))) _Float16 v16h;
typedef __attribute__((ext_vector_type(8)))  _Float16 h8;
typedef __attribute__((ext_vector_type(4)))  _Float16 h4;
typedef __attribute__((ext_vector_type(8)))  float    v8f;
typedef __attribute__((ext_vector_type(4)))  int      v4i;

union V16 { v16h v; h8 h[2]; };

#define WMMA_F16(a, b, c) \
  __builtin_amdgcn_wmma_f32_16x16x32_f16(false, (a), false, (b), (short)0, (c), false, false)

// ---- async global->LDS (16B per lane), with sync fallback --------------------
#if defined(__has_builtin)
#if __has_builtin(__builtin_amdgcn_global_load_async_to_lds_b128)
#define HAVE_ASYNC_LDS 1
#endif
#endif
#ifndef HAVE_ASYNC_LDS
#define HAVE_ASYNC_LDS 0
#endif

__device__ __forceinline__ void copy16_to_lds(_Float16* lds_dst,
                                              const _Float16* gsrc) {
#if HAVE_ASYNC_LDS
  __builtin_amdgcn_global_load_async_to_lds_b128(
      (__attribute__((address_space(1))) v4i*)gsrc,
      (__attribute__((address_space(3))) v4i*)lds_dst,
      /*imm offset*/ 0, /*cpol*/ 0);
#else
  *(h8*)lds_dst = *(const h8*)gsrc;
#endif
}

__device__ __forceinline__ void wait_async_lds() {
#if HAVE_ASYNC_LDS
#if defined(__has_builtin) && __has_builtin(__builtin_amdgcn_s_wait_asynccnt)
  __builtin_amdgcn_s_wait_asynccnt(0);
#else
  asm volatile("s_wait_asynccnt 0" ::: "memory");
#endif
#endif
}

// Problem constants (match the reference).
constexpr int kB = 2, kN = 2048, kC = 1024, kH = 16, kD = 64;

// ---------------------------------------------------------------------------
// fp32 -> fp16 cast (vectorized 4-wide; all sizes are multiples of 4)
// ---------------------------------------------------------------------------
__global__ void cast_f32_f16(const float* __restrict__ src,
                             _Float16* __restrict__ dst, int n) {
  int i = (blockIdx.x * blockDim.x + threadIdx.x) * 4;
  int stride = gridDim.x * blockDim.x * 4;
  for (; i < n; i += stride) {
    float4 f = *(const float4*)(src + i);
    h4 o = { (_Float16)f.x, (_Float16)f.y, (_Float16)f.z, (_Float16)f.w };
    *(h4*)(dst + i) = o;
  }
}

// ---------------------------------------------------------------------------
// WMMA GEMM:  out(MxN) = A(MxK, f16 row-major) @ Bw(KxN, f16 row-major) + bias
// Block = 256 threads (8 waves). Block tile 128x128; wave tile 32x64
// (2 m-subtiles x 4 n-tiles = 8 accumulator tiles). K stepped by 32,
// double-buffered: async A fill + B global fetch one step ahead, B transpose
// stores after the WMMAs, single barrier per step.
// ---------------------------------------------------------------------------
template <bool OUT16>
__global__ void gemm_wmma_kernel(const _Float16* __restrict__ A,
                                 const _Float16* __restrict__ Bw,
                                 const float* __restrict__ bias,
                                 void* __restrict__ outp,
                                 int M, int N, int K) {
  __shared__ _Float16 Atile[2][128 * 32];   // [m][k]
  __shared__ _Float16 BtileT[2][128 * 32];  // [n][k]

  const int tid = threadIdx.x;
  const int w   = tid >> 5;   // wave 0..7
  const int l   = tid & 31;
  const int hf  = l >> 4;     // lane half
  const int ln  = l & 15;
  const int wvm = w & 3;      // 32-row strip within block tile
  const int wvn = w >> 2;     // 64-col strip within block tile
  const int m0  = blockIdx.y * 128;
  const int n0  = blockIdx.x * 128;

  // fill mappings
  const int arow = tid >> 1, akc = (tid & 1) * 16;            // A: 2 x b128
  const int bk0 = (tid * 2) >> 4, bn0 = ((tid * 2) & 15) * 8; // B chunk 0
  const int bk1 = (tid * 2 + 1) >> 4, bn1 = ((tid * 2 + 1) & 15) * 8;

  auto fill_a = [&](int buf, int k0) {
    const _Float16* ap = &A[(size_t)(m0 + arow) * K + k0 + akc];
    copy16_to_lds(&Atile[buf][arow * 32 + akc], ap);
    copy16_to_lds(&Atile[buf][arow * 32 + akc + 8], ap + 8);
  };
  auto load_b = [&](int k0, h8* bv) {
    bv[0] = *(const h8*)&Bw[(size_t)(k0 + bk0) * N + n0 + bn0];
    bv[1] = *(const h8*)&Bw[(size_t)(k0 + bk1) * N + n0 + bn1];
  };
  auto store_bT = [&](int buf, const h8* bv) {
#pragma unroll
    for (int i = 0; i < 8; ++i) BtileT[buf][(bn0 + i) * 32 + bk0] = bv[0][i];
#pragma unroll
    for (int i = 0; i < 8; ++i) BtileT[buf][(bn1 + i) * 32 + bk1] = bv[1][i];
  };

  v8f acc[2][4] = {};

  // ---- prologue: fill buffer 0 ----
  {
    h8 bv[2];
    fill_a(0, 0);
    load_b(0, bv);
    store_bT(0, bv);
  }
  wait_async_lds();
  __syncthreads();

  const int nsteps = K / 32;
  for (int it = 0; it < nsteps; ++it) {
    const int cur = it & 1, nxt = cur ^ 1;
    const bool have_next = (it + 1 < nsteps);
    h8 bv[2];
    if (have_next) {              // issue next tile's traffic before compute
      fill_a(nxt, (it + 1) * 32);
      load_b((it + 1) * 32, bv);
    }

    // ---- fragments + WMMA on current buffer ----
    V16 a[2];  // A 16x32: halves i -> k = (i%8) + 16*(i/8) + 8*hf
#pragma unroll
    for (int u = 0; u < 2; ++u) {
      int m = wvm * 32 + u * 16 + ln;
      a[u].h[0] = *(const h8*)&Atile[cur][m * 32 + hf * 8];
      a[u].h[1] = *(const h8*)&Atile[cur][m * 32 + hf * 8 + 16];
    }
#pragma unroll
    for (int t = 0; t < 4; ++t) {
      V16 b;  // B 32x16: halves j -> k = (j%8) + 16*(j/8) + 8*hf, n = ln
      int n = wvn * 64 + t * 16 + ln;
      b.h[0] = *(const h8*)&BtileT[cur][n * 32 + hf * 8];
      b.h[1] = *(const h8*)&BtileT[cur][n * 32 + hf * 8 + 16];
      acc[0][t] = WMMA_F16(a[0].v, b.v, acc[0][t]);
      acc[1][t] = WMMA_F16(a[1].v, b.v, acc[1][t]);
    }

    if (have_next) store_bT(nxt, bv);  // transpose stores after the math
    wait_async_lds();
    __syncthreads();
  }

  // ---- bias + store (C layout: VGPR r -> m = 8*hf + r, n = ln) ----
#pragma unroll
  for (int t = 0; t < 4; ++t) {
    int n = n0 + wvn * 64 + t * 16 + ln;
    float bv = bias[n];
#pragma unroll
    for (int u = 0; u < 2; ++u) {
#pragma unroll
      for (int r = 0; r < 8; ++r) {
        int m = m0 + wvm * 32 + u * 16 + hf * 8 + r;
        float val = acc[u][t][r] + bv;
        if (OUT16)
          ((_Float16*)outp)[(size_t)m * N + n] = (_Float16)val;
        else
          ((float*)outp)[(size_t)m * N + n] = val;
      }
    }
  }
}

// ---------------------------------------------------------------------------
// Flash attention: one block = 128 query rows of one (b,h); 256 threads
// (8 waves x 16 query rows). 64-key K/V tiles double-buffered through LDS:
// async K fill + V global fetch issued one tile ahead, V transpose stores
// after the WMMAs. Online softmax row stats live in registers (shfl_xor
// masks 1..8 stay inside each 16-lane half, matching the WMMA C layout).
// qkv layout: (B, N, 3, H, D) -> row stride 3C; Q at +0, K at +C, V at +2C.
// ---------------------------------------------------------------------------
__global__ void attn_wmma_kernel(const _Float16* __restrict__ qkv,
                                 _Float16* __restrict__ out) {
  __shared__ _Float16 Ktile[2][64 * 64];    // [key][d] (contig d = contig k for Q*K^T)
  __shared__ _Float16 VtileT[2][64 * 64];   // [d][key] (contig key = contig k for P*V)
  __shared__ _Float16 Ptile[8 * 16 * 64];   // per-wave P, [m][key]

  const int tid = threadIdx.x;  // 0..255
  const int w   = tid >> 5;     // wave 0..7
  const int l   = tid & 31;
  const int hf  = l >> 4;
  const int ln  = l & 15;

  const int bh = blockIdx.x;    // 0..B*H-1
  const int b  = bh / kH;
  const int h  = bh % kH;
  const int q0 = blockIdx.y * 128;

  const size_t rs = 3 * kC;  // 3072
  const _Float16* qbase = qkv + (size_t)(b * kN) * rs + h * kD;
  const _Float16* kbase = qbase + kC;
  const _Float16* vbase = qbase + 2 * kC;

  // fill mapping: 4 threads per key row, 16 halves each
  const int frow = tid >> 2, fdc = (tid & 3) * 16;

  auto fill_k = [&](int buf, int kb) {
    const _Float16* kp = kbase + (size_t)(kb * 64 + frow) * rs + fdc;
    copy16_to_lds(&Ktile[buf][frow * 64 + fdc], kp);
    copy16_to_lds(&Ktile[buf][frow * 64 + fdc + 8], kp + 8);
  };
  auto load_v = [&](int kb, h8* vv) {
    const _Float16* vp = vbase + (size_t)(kb * 64 + frow) * rs + fdc;
    vv[0] = *(const h8*)&vp[0];
    vv[1] = *(const h8*)&vp[8];
  };
  auto store_vT = [&](int buf, const h8* vv) {
#pragma unroll
    for (int c = 0; c < 2; ++c)
#pragma unroll
      for (int i = 0; i < 8; ++i)
        VtileT[buf][(fdc + c * 8 + i) * 64 + frow] = vv[c][i];
  };

  // Q fragments (pre-scaled by 1/sqrt(D)=0.125, exact in f16); fixed all kernel
  V16 qf[2];
  {
    int qrow = q0 + w * 16 + ln;
    const _Float16* qp = qbase + (size_t)qrow * rs;
#pragma unroll
    for (int s = 0; s < 2; ++s) {
      h8 lo = *(const h8*)&qp[s * 32 + hf * 8];
      h8 hi = *(const h8*)&qp[s * 32 + hf * 8 + 16];
#pragma unroll
      for (int i = 0; i < 8; ++i) {
        lo[i] = lo[i] * (_Float16)0.125f;
        hi[i] = hi[i] * (_Float16)0.125f;
      }
      qf[s].h[0] = lo;
      qf[s].h[1] = hi;
    }
  }

  v8f ot[4] = {};        // O accumulator, 16x64 f32
  float mrow[8], lrow[8];
#pragma unroll
  for (int r = 0; r < 8; ++r) { mrow[r] = -1e30f; lrow[r] = 0.f; }

  _Float16* myP = &Ptile[w * 16 * 64];

  // ---- prologue: fill buffer 0 with key-block 0 ----
  {
    h8 vv[2];
    fill_k(0, 0);
    load_v(0, vv);
    store_vT(0, vv);
  }
  wait_async_lds();
  __syncthreads();

  const int nkb = kN / 64;
  for (int kb = 0; kb < nkb; ++kb) {
    const int cur = kb & 1, nxt = cur ^ 1;
    const bool have_next = (kb + 1 < nkb);
    h8 vv[2];
    if (have_next) {              // issue next tile's traffic before compute
      fill_k(nxt, kb + 1);
      load_v(kb + 1, vv);
    }

    // ---- S = (Q/8) @ K^T : 16x64 per wave, 8 WMMAs ----
    v8f st[4] = {};
#pragma unroll
    for (int s = 0; s < 2; ++s) {
#pragma unroll
      for (int t = 0; t < 4; ++t) {
        V16 bf;  // B[k=d][n=key]: j -> d = s*32 + 8*hf + (j%8) + 16*(j/8)
        int key = t * 16 + ln;
        bf.h[0] = *(const h8*)&Ktile[cur][key * 64 + s * 32 + hf * 8];
        bf.h[1] = *(const h8*)&Ktile[cur][key * 64 + s * 32 + hf * 8 + 16];
        st[t] = WMMA_F16(qf[s].v, bf.v, st[t]);
      }
    }

    // ---- online softmax; row r lives at m = 8*hf + r, spread over 16 lanes ----
#pragma unroll
    for (int r = 0; r < 8; ++r) {
      float vmax = fmaxf(fmaxf(st[0][r], st[1][r]), fmaxf(st[2][r], st[3][r]));
#pragma unroll
      for (int sh = 1; sh <= 8; sh <<= 1)
        vmax = fmaxf(vmax, __shfl_xor(vmax, sh, 32));
      float mnew  = fmaxf(mrow[r], vmax);
      float scale = __expf(mrow[r] - mnew);
      mrow[r] = mnew;
      float psum = 0.f;
#pragma unroll
      for (int t = 0; t < 4; ++t) {
        float p = __expf(st[t][r] - mnew);
        psum += p;
        myP[(hf * 8 + r) * 64 + t * 16 + ln] = (_Float16)p;  // C-layout -> [m][key]
      }
#pragma unroll
      for (int sh = 1; sh <= 8; sh <<= 1) psum += __shfl_xor(psum, sh, 32);
      lrow[r] = lrow[r] * scale + psum;
#pragma unroll
      for (int t = 0; t < 4; ++t) ot[t][r] *= scale;  // rescale O accumulator
    }

    // Ptile is wave-private: only our own DS stores must land before our reads.
    asm volatile("s_wait_dscnt 0" ::: "memory");

    // ---- O += P @ V : 8 WMMAs ----
#pragma unroll
    for (int s = 0; s < 2; ++s) {
      V16 pa;  // A 16x32 over keys: i -> key = s*32 + 8*hf + (i%8) + 16*(i/8)
      pa.h[0] = *(const h8*)&myP[ln * 64 + s * 32 + hf * 8];
      pa.h[1] = *(const h8*)&myP[ln * 64 + s * 32 + hf * 8 + 16];
#pragma unroll
      for (int t = 0; t < 4; ++t) {
        V16 bf;  // B[k=key][n=d]: j -> key = s*32 + 8*hf + (j%8) + 16*(j/8)
        int d = t * 16 + ln;
        bf.h[0] = *(const h8*)&VtileT[cur][d * 64 + s * 32 + hf * 8];
        bf.h[1] = *(const h8*)&VtileT[cur][d * 64 + s * 32 + hf * 8 + 16];
        ot[t] = WMMA_F16(pa.v, bf.v, ot[t]);
      }
    }

    if (have_next) store_vT(nxt, vv);  // transpose stores after the math
    wait_async_lds();
    __syncthreads();
  }

  // ---- normalize, write attention output (B,N,H*D) as f16 ----
#pragma unroll
  for (int r = 0; r < 8; ++r) {
    float inv = 1.0f / lrow[r];
    int qrow = q0 + w * 16 + hf * 8 + r;
    _Float16* op = out + (size_t)(b * kN + qrow) * kC + h * kD;
#pragma unroll
    for (int t = 0; t < 4; ++t)
      op[t * 16 + ln] = (_Float16)(ot[t][r] * inv);
  }
}

// ---------------------------------------------------------------------------
// Host-side orchestration (graph-capture safe: kernels only, all on stream)
// ---------------------------------------------------------------------------
extern "C" void kernel_launch(void* const* d_in, const int* in_sizes, int n_in,
                              void* d_out, int out_size, void* d_ws,
                              size_t ws_size, hipStream_t stream) {
  (void)in_sizes; (void)n_in; (void)out_size; (void)ws_size;

  const float* x      = (const float*)d_in[0];  // (B,N,C)
  const float* qkv_w  = (const float*)d_in[1];  // (C,3C)
  const float* qkv_b  = (const float*)d_in[2];  // (3C,)
  const float* proj_w = (const float*)d_in[3];  // (C,C)
  const float* proj_b = (const float*)d_in[4];  // (C,)
  float* out = (float*)d_out;                   // (B,N,C) f32

  // Workspace layout (f16 halves), ~48 MB total
  char* ws = (char*)d_ws;
  _Float16* xh    = (_Float16*)(ws);                             //  8.0 MB
  _Float16* wqkvh = (_Float16*)(ws + (size_t)8  * 1024 * 1024);  //  6.0 MB
  _Float16* wprjh = (_Float16*)(ws + (size_t)14 * 1024 * 1024);  //  2.0 MB
  _Float16* qkvh  = (_Float16*)(ws + (size_t)16 * 1024 * 1024);  // 24.0 MB
  _Float16* attnh = (_Float16*)(ws + (size_t)40 * 1024 * 1024);  //  8.0 MB

  const int nX  = kB * kN * kC;  // 4194304
  const int nWq = kC * 3 * kC;   // 3145728
  const int nWp = kC * kC;       // 1048576

  cast_f32_f16<<<1024, 256, 0, stream>>>(x, xh, nX);
  cast_f32_f16<<<1024, 256, 0, stream>>>(qkv_w, wqkvh, nWq);
  cast_f32_f16<<<1024, 256, 0, stream>>>(proj_w, wprjh, nWp);

  // QKV projection: (4096 x 1024) @ (1024 x 3072) + b -> f16
  gemm_wmma_kernel<true><<<dim3(3 * kC / 128, kB * kN / 128), 256, 0, stream>>>(
      xh, wqkvh, qkv_b, qkvh, kB * kN, 3 * kC, kC);

  // Flash attention over all (b,h), 128 query rows per block
  attn_wmma_kernel<<<dim3(kB * kH, kN / 128), 256, 0, stream>>>(qkvh, attnh);

  // Output projection: (4096 x 1024) @ (1024 x 1024) + b -> f32
  gemm_wmma_kernel<false><<<dim3(kC / 128, kB * kN / 128), 256, 0, stream>>>(
      attnh, wprjh, proj_b, out, kB * kN, kC, kC);
}
